// DeformableSpatialAttentionLayer3D_9156870275453
// MI455X (gfx1250) — matrix operand order repaired and checked by reading, not compiled, not run on hardware
//
#include <hip/hip_runtime.h>
#include <hip/hip_bf16.h>

// ---------------------------------------------------------------------------
// DeformableSpatialAttentionLayer3D for MI455X (gfx1250, wave32, WMMA)
// Fixed problem sizes from the reference setup.
// ---------------------------------------------------------------------------
#define BS     2
#define HGT    64
#define WID    80
#define NQ     (HGT * WID)          // 5120
#define E_DIM  576
#define HEADS  18
#define NPTS   12
#define DD     5                    // depth of value volume
#define HD     32                   // head dim
#define NTOK   (NQ * DD)            // 25600 tokens per batch
#define NOFF   (HEADS * NPTS * 3)   // 648
#define NATT   (HEADS * NPTS)       // 216
#define NSCORE (NOFF + NATT)        // 864

typedef __attribute__((ext_vector_type(16))) __bf16 v16bf;
typedef __attribute__((ext_vector_type(8)))  float  v8f;

// ---------------------------------------------------------------------------
// Fused GEMM:  D = (A [+ Aadd]) @ [B1 | B2] + [bias1 | bias2]
//   A: MxK fp32 row-major (lda = K), M % 64 == 0, K % 32 == 0.
//   B1: K x nsplit, B2: K x (Ntot - nsplit); split point % 4 == 0.
// Computed in bf16 via V_WMMA_F32_16X16X32_BF16 with f32 accumulation.
// Block: 128 threads (4 waves). Tile: 64M x 64N, K step 32.
// Register-staged software pipeline: global float4 loads for tile k+1
// overlap WMMA on tile k; single LDS buffer.
// ---------------------------------------------------------------------------
template <bool HAS_ADD, bool SPLITN>
__global__ __launch_bounds__(128)
void dsal3d_gemm_bf16(const float* __restrict__ A,
                      const float* __restrict__ Aadd,
                      const float* __restrict__ B1,
                      const float* __restrict__ B2,
                      const float* __restrict__ bias1,
                      const float* __restrict__ bias2,
                      float* __restrict__ D,
                      int M, int Ntot, int nsplit, int K,
                      int ldb1, int ldb2, int ldd)
{
    // LDS tiles. Stride 40 bf16 (80 B) keeps every fragment chunk 16B aligned.
    __shared__ __align__(16) __bf16 lsA[64 * 40];   // row-major: [m][k]
    __shared__ __align__(16) __bf16 lsB[64 * 40];   // N-major:   [n][k]

    const int t     = threadIdx.x;        // 0..127
    const int wv    = t >> 5;             // wave 0..3
    const int lane  = t & 31;
    const int mBase = blockIdx.x * 64;
    const int nBase = blockIdx.y * 64;

    v8f acc[4] = {};                      // wave strip: rows wv*16..+15, cols 0..63

    // Fragment addressing (per CDNA5 ISA 16-bit WMMA layouts):
    //  A (16x32): lane L holds row M=L&15; lanes<16: K 0..7 & 16..23,
    //             lanes>=16: K 8..15 & 24..31  -> two 16B LDS chunks.
    //  B (32x16): lane L holds col N=L&15; lanes<16: K 0..7 & 16..23,
    //             lanes>=16: K 8..15 & 24..31 -> two 16B LDS chunks.
    const int arow = (wv << 4) + (lane & 15);
    const int akb  = (lane >> 4) << 3;          // 0 or 8
    const int bcol = lane & 15;
    const int bkb  = (lane >> 4) << 3;          // 0 or 8

    // Per-thread tile-load decomposition (all float4, all 16B aligned):
    //  A tile 64x32: group g = t + i*128 (i<4): row g>>3, k = (g&7)*4
    //  B tile 32x64: group g = t + i*128 (i<4): k = g>>4, n = (g&15)*4
    float4 ra[4], rb[4];

    auto load_tiles = [&](int kb) {
        #pragma unroll
        for (int i = 0; i < 4; ++i) {
            int g = t + (i << 7);
            int r = g >> 3;
            int k = (g & 7) << 2;
            const float* p = A + (size_t)(mBase + r) * K + (kb + k);
            float4 v = *(const float4*)p;
            if (HAS_ADD) {
                const float* p2 = Aadd + (size_t)(mBase + r) * K + (kb + k);
                float4 w = *(const float4*)p2;
                v.x += w.x; v.y += w.y; v.z += w.z; v.w += w.w;
            }
            ra[i] = v;
            if (kb + 64 < K) __builtin_prefetch(p + 32, 0, 1);
        }
        #pragma unroll
        for (int i = 0; i < 4; ++i) {
            int g  = t + (i << 7);
            int k  = g >> 4;
            int n4 = (g & 15) << 2;
            int gn = nBase + n4;
            float4 v;
            if (!SPLITN) {
                v = *(const float4*)(B1 + (size_t)(kb + k) * ldb1 + gn);
            } else {
                // split point and Ntot are multiples of 4: groups never straddle
                if (gn < nsplit) {
                    v = *(const float4*)(B1 + (size_t)(kb + k) * ldb1 + gn);
                } else if (gn < Ntot) {
                    v = *(const float4*)(B2 + (size_t)(kb + k) * ldb2 + (gn - nsplit));
                } else {
                    v = make_float4(0.f, 0.f, 0.f, 0.f);
                }
            }
            rb[i] = v;
        }
    };

    auto store_tiles = [&]() {
        #pragma unroll
        for (int i = 0; i < 4; ++i) {
            int g = t + (i << 7);
            int r = g >> 3;
            int k = (g & 7) << 2;
            union { __bf16 h[4]; unsigned long long u; } pk;
            pk.h[0] = (__bf16)ra[i].x;
            pk.h[1] = (__bf16)ra[i].y;
            pk.h[2] = (__bf16)ra[i].z;
            pk.h[3] = (__bf16)ra[i].w;
            *(unsigned long long*)(lsA + r * 40 + k) = pk.u;   // one ds_store_b64
        }
        #pragma unroll
        for (int i = 0; i < 4; ++i) {
            int g  = t + (i << 7);
            int k  = g >> 4;
            int n4 = (g & 15) << 2;
            lsB[(n4 + 0) * 40 + k] = (__bf16)rb[i].x;          // transposed scatter
            lsB[(n4 + 1) * 40 + k] = (__bf16)rb[i].y;
            lsB[(n4 + 2) * 40 + k] = (__bf16)rb[i].z;
            lsB[(n4 + 3) * 40 + k] = (__bf16)rb[i].w;
        }
    };

    load_tiles(0);

    for (int kb = 0; kb < K; kb += 32) {
        __syncthreads();          // previous iteration's readers done
        store_tiles();
        __syncthreads();

        if (kb + 32 < K) load_tiles(kb + 32);   // overlaps with WMMA below

        union { v16bf v; float4 f[2]; } ua;
        ua.f[0] = *(const float4*)(lsA + arow * 40 + akb);
        ua.f[1] = *(const float4*)(lsA + arow * 40 + akb + 16);

        #pragma unroll
        for (int nn = 0; nn < 4; ++nn) {
            union { v16bf v; float4 f[2]; } ub;
            const __bf16* bp = lsB + (nn * 16 + bcol) * 40 + bkb;
            ub.f[0] = *(const float4*)(bp);
            ub.f[1] = *(const float4*)(bp + 16);
            acc[nn] = __builtin_amdgcn_wmma_f32_16x16x32_bf16(
                false, ua.v, false, ub.v, (short)0, acc[nn], false, false);
        }
    }

    // ---- epilogue: bias + store ------------------------------------------
    #pragma unroll
    for (int nn = 0; nn < 4; ++nn) {
        int gn = nBase + nn * 16 + (lane & 15);
        float bs;
        if (!SPLITN) {
            bs = bias1[gn];
        } else {
            if (gn >= Ntot) continue;
            bs = (gn < nsplit) ? bias1[gn] : bias2[gn - nsplit];
        }
        int mrow = mBase + (wv << 4) + ((lane >> 4) << 3);
        #pragma unroll
        for (int j = 0; j < 8; ++j) {
            D[(size_t)(mrow + j) * ldd + gn] = acc[nn][j] + bs;
        }
    }
}

// ---------------------------------------------------------------------------
// Trilinear deformable sampling + softmax over points.
// One wave per (b, q, head); lane = head-dim channel (coalesced 128B gathers).
// ---------------------------------------------------------------------------
__global__ __launch_bounds__(256)
void dsal3d_sample(const float* __restrict__ vproj,   // (BS*NTOK, E)
                   const float* __restrict__ scores,  // (BS*NQ, NSCORE)
                   float* __restrict__ interm)        // (BS*NQ, E)
{
    const int wv   = threadIdx.x >> 5;
    const int lane = threadIdx.x & 31;
    const int g    = blockIdx.x * 8 + wv;             // (b,q,head) group
    if (g >= BS * NQ * HEADS) return;

    const int b    = g / (NQ * HEADS);
    const int rem  = g - b * (NQ * HEADS);
    const int q    = rem / HEADS;
    const int hidx = rem - q * HEADS;

    const float* srow = scores + (size_t)(b * NQ + q) * NSCORE;

    float lg[NPTS], ox[NPTS], oy[NPTS], oz[NPTS];
    #pragma unroll
    for (int p = 0; p < NPTS; ++p) {
        lg[p] = srow[NOFF + hidx * NPTS + p];
        ox[p] = srow[hidx * 36 + p * 3 + 0];
        oy[p] = srow[hidx * 36 + p * 3 + 1];
        oz[p] = srow[hidx * 36 + p * 3 + 2];
    }

    // softmax over the 12 points
    float mx = lg[0];
    #pragma unroll
    for (int p = 1; p < NPTS; ++p) mx = fmaxf(mx, lg[p]);
    float ssum = 0.0f;
    #pragma unroll
    for (int p = 0; p < NPTS; ++p) { lg[p] = __expf(lg[p] - mx); ssum += lg[p]; }
    const float sinv = 1.0f / ssum;

    // reference point: pixel center, z = 0; q = y*W + x
    const int qy = q / WID;
    const int qx = q - qy * WID;
    const float refx = ((float)qx + 0.5f);
    const float refy = ((float)qy + 0.5f);

    const float* vb = vproj + (size_t)b * NTOK * E_DIM + hidx * HD + lane;
    float acc = 0.0f;

    #pragma unroll
    for (int p = 0; p < NPTS; ++p) {
        // pix = u*size - 0.5 ; norm = (W, H, NF=3), z scaled by D/NF = 5/3
        float x = refx + ox[p] - 0.5f;
        float y = refy + oy[p] - 0.5f;
        float z = oz[p] * (5.0f / 3.0f) - 0.5f;
        float x0 = floorf(x), y0 = floorf(y), z0 = floorf(z);
        float fx = x - x0, fy = y - y0, fz = z - z0;
        int ix0 = (int)x0, iy0 = (int)y0, iz0 = (int)z0;
        float wp = lg[p] * sinv;

        #pragma unroll
        for (int dz = 0; dz < 2; ++dz) {
            float wz = dz ? fz : (1.0f - fz);
            int zi = iz0 + dz;
            int zc = min(max(zi, 0), DD - 1);
            bool vz = (zi >= 0) && (zi < DD);
            #pragma unroll
            for (int dy = 0; dy < 2; ++dy) {
                float wy = dy ? fy : (1.0f - fy);
                int yi = iy0 + dy;
                int yc = min(max(yi, 0), HGT - 1);
                bool vy = (yi >= 0) && (yi < HGT);
                #pragma unroll
                for (int dx = 0; dx < 2; ++dx) {
                    float wx = dx ? fx : (1.0f - fx);
                    int xi = ix0 + dx;
                    int xc = min(max(xi, 0), WID - 1);
                    bool vv = vz && vy && (xi >= 0) && (xi < WID);
                    int tok = (zc * HGT + yc) * WID + xc;
                    float gv = vb[(size_t)tok * E_DIM];      // coalesced 128B/wave
                    acc += wp * (vv ? (wz * wy * wx) : 0.0f) * gv;
                }
            }
        }
    }

    interm[(size_t)(b * NQ + q) * E_DIM + hidx * HD + lane] = acc;
}

// ---------------------------------------------------------------------------
// Host-side launcher.
// Inputs (setup_inputs order): 0 query, 1 value, 2 query_pos, 3 W_off,
// 4 b_off, 5 W_attn, 6 b_attn, 7 W_v, 8 b_v, 9 W_out, 10 b_out, 11 h, 12 w
// ---------------------------------------------------------------------------
extern "C" void kernel_launch(void* const* d_in, const int* in_sizes, int n_in,
                              void* d_out, int out_size, void* d_ws, size_t ws_size,
                              hipStream_t stream)
{
    const float* query     = (const float*)d_in[0];
    const float* value     = (const float*)d_in[1];
    const float* query_pos = (const float*)d_in[2];
    const float* W_off     = (const float*)d_in[3];
    const float* b_off     = (const float*)d_in[4];
    const float* W_attn    = (const float*)d_in[5];
    const float* b_attn    = (const float*)d_in[6];
    const float* W_v       = (const float*)d_in[7];
    const float* b_v       = (const float*)d_in[8];
    const float* W_out     = (const float*)d_in[9];
    const float* b_out     = (const float*)d_in[10];
    float* out = (float*)d_out;

    // workspace carve
    float* vproj  = (float*)d_ws;                                  // 51200 x 576
    float* scores = vproj + (size_t)BS * NTOK * E_DIM;             // 10240 x 864
    float* interm = scores + (size_t)BS * NQ * NSCORE;             // 10240 x 576

    // 1) value projection: (BS*NTOK, 576) @ (576, 576) + b_v
    {
        const int M = BS * NTOK;        // 51200
        dim3 grid(M / 64, E_DIM / 64);  // (800, 9)
        dsal3d_gemm_bf16<false, false><<<grid, 128, 0, stream>>>(
            value, nullptr, W_v, nullptr, b_v, nullptr, vproj,
            M, E_DIM, E_DIM, E_DIM, E_DIM, E_DIM, E_DIM);
    }

    // 2) fused offsets+attn logits: (query+query_pos) @ [W_off | W_attn]
    {
        const int M = BS * NQ;          // 10240
        dim3 grid(M / 64, (NSCORE + 63) / 64);   // (160, 14)
        dsal3d_gemm_bf16<true, true><<<grid, 128, 0, stream>>>(
            query, query_pos, W_off, W_attn, b_off, b_attn, scores,
            M, NSCORE, NOFF, E_DIM, NOFF, NATT, NSCORE);
    }

    // 3) softmax + trilinear deformable sampling
    {
        const int groups = BS * NQ * HEADS;      // 184320
        dsal3d_sample<<<groups / 8, 256, 0, stream>>>(vproj, scores, interm);
    }

    // 4) output projection: interm @ W_out + b_out -> d_out
    {
        const int M = BS * NQ;          // 10240
        dim3 grid(M / 64, E_DIM / 64);  // (160, 9)
        dsal3d_gemm_bf16<false, false><<<grid, 128, 0, stream>>>(
            interm, nullptr, W_out, nullptr, b_out, nullptr, out,
            M, E_DIM, E_DIM, E_DIM, E_DIM, E_DIM, E_DIM);
    }
}
